// TwoMLPHead_25709674234127
// MI455X (gfx1250) — matrix operand order repaired and checked by reading, not compiled, not run on hardware
//
#include <hip/hip_runtime.h>
#include <hip/hip_bf16.h>
#include <math.h>

typedef __bf16 bf16_t;
typedef __bf16 v16bf __attribute__((ext_vector_type(16)));
typedef float  v8f   __attribute__((ext_vector_type(8)));

#define NTOT   2048
#define FEAT   1024
#define GRP    16
#define DGRP   64
#define EMBD   64
#define INCH   12544
#define NGT    300
#define MPAD   320

// ---------------------------------------------------------------------------
// Async global->LDS copy of 32 bytes per lane (2 x b128), no VGPR round-trip.
// INST_OFFSET applies to both LDS and global addresses (ISA 08_async_tensor).
// Tracked by ASYNCcnt.
// ---------------------------------------------------------------------------
__device__ __forceinline__ void async_cp32(unsigned lds_off, const void* gptr)
{
    asm volatile("global_load_async_to_lds_b128 %0, %1, off\n\t"
                 "global_load_async_to_lds_b128 %0, %1, off offset:16"
                 :: "v"(lds_off), "v"(gptr) : "memory");
}

__device__ __forceinline__ void wait_async0()
{
    asm volatile("s_wait_asynccnt 0" ::: "memory");
}

// load one 32B WMMA B-fragment chunk pair from LDS
__device__ __forceinline__ v16bf load_frag(const bf16_t* p0, const bf16_t* p1)
{
    v16bf f;
    *((uint4*)&f + 0) = *(const uint4*)p0;
    *((uint4*)&f + 1) = *(const uint4*)p1;
    return f;
}

// ---------------------------------------------------------------------------
// Batched bf16 WMMA GEMM:  C = act( alpha * A(MxK) @ Bt(NxK)^T + bias + res )
// Block tile 64 x (64*CB) x 32, 128 threads = 4 waves; wave w owns rows
// [16w,16w+16) and all 64*CB columns -> 4*CB WMMA per k-step per wave with a
// single A-fragment. Double-buffered LDS fed by async global->LDS DMA, and
// software-pipelined B-fragment loads so ds_load overlaps the in-flight WMMA.
// Requires M%64==0, N%(64*CB)==0, K%32==0 (guaranteed by caller).
// ---------------------------------------------------------------------------
template<int CB>
__global__ __launch_bounds__(128)
void gemm_bf16_wmma(const bf16_t* __restrict__ A, long batA, int lda,
                    const bf16_t* __restrict__ Bt, long batB, int ldb,
                    const float* __restrict__ bias, long batBias,
                    const float* __restrict__ res, long batRes, int ldres,
                    float* __restrict__ Cf, long batCf, int ldcf,
                    bf16_t* __restrict__ Cb, long batCb, int ldcb,
                    int K, float alpha, int relu)
{
    __shared__ __attribute__((aligned(16))) bf16_t sA[2][64 * 32];
    __shared__ __attribute__((aligned(16))) bf16_t sB[2][64 * CB * 32];

    const int z = blockIdx.z;
    A  += (long)z * batA;
    Bt += (long)z * batB;
    if (bias) bias += (long)z * batBias;
    if (res)  res  += (long)z * batRes;
    if (Cf)   Cf   += (long)z * batCf;
    if (Cb)   Cb   += (long)z * batCb;

    const int m0 = blockIdx.y * 64;
    const int n0 = blockIdx.x * (64 * CB);
    const int t    = threadIdx.x;
    const int wid  = t >> 5;
    const int lane = t & 31;
    const int fr   = lane & 15;   // row(A)/col(B) index inside 16
    const int sel  = lane >> 4;   // K-half selector

    // staging: thread t copies the 16-element half (ldh) of tile row ldr
    const int ldr = t >> 1;       // 0..63
    const int ldh = t & 1;

    auto stage = [&](int kk, int b) {
        async_cp32((unsigned)(size_t)&sA[b][ldr * 32 + ldh * 16],
                   A + (size_t)(m0 + ldr) * lda + kk + ldh * 16);
        #pragma unroll
        for (int r = 0; r < CB; ++r) {
            const int rb = ldr + r * 64;
            async_cp32((unsigned)(size_t)&sB[b][rb * 32 + ldh * 16],
                       Bt + (size_t)(n0 + rb) * ldb + kk + ldh * 16);
        }
    };

    v8f c[4 * CB] = {};

    stage(0, 0);
    for (int kk = 0, i = 0; kk < K; kk += 32, ++i) {
        const int cur = i & 1;
        wait_async0();        // this wave's tile-i DMA into LDS complete
        __syncthreads();      // all waves' tiles published
        if (kk + 32 < K) stage(kk + 32, cur ^ 1);  // overlaps compute below

        // A fragment (16x32): lane holds 16 contiguous K at sel*16 of row fr
        const bf16_t* pa = &sA[cur][(wid * 16 + fr) * 32];
        const v16bf af = load_frag(pa + sel * 16, pa + sel * 16 + 8);

        // B fragments (32x16): lane holds 16 contiguous K at sel*16 of col fr.
        // Software pipeline: fetch b[j+1] while wmma j executes.
        const bf16_t* pb0 = &sB[cur][fr * 32 + sel * 16];
        v16bf bcur = load_frag(pb0, pb0 + 8);
        #pragma unroll
        for (int j = 0; j < 4 * CB; ++j) {
            v16bf bnxt;
            if (j + 1 < 4 * CB) {
                const bf16_t* pb = &sB[cur][((j + 1) * 16 + fr) * 32 + sel * 16];
                bnxt = load_frag(pb, pb + 8);
            }
            c[j] = __builtin_amdgcn_wmma_f32_16x16x32_bf16(
                       false, af, false, bcur, (short)0, c[j], false, false);
            if (j + 1 < 4 * CB) bcur = bnxt;
        }
    }

    // Epilogue: C lane map — lane<16: (M=r, N=lane); lane>=16: (M=r+8, N=lane-16)
    const int row0 = m0 + wid * 16 + sel * 8;
    #pragma unroll
    for (int j = 0; j < 4 * CB; ++j) {
        const int col = n0 + j * 16 + fr;
        const float bv = bias ? bias[col] : 0.f;
        #pragma unroll
        for (int r = 0; r < 8; ++r) {
            const int row = row0 + r;
            float v = alpha * c[j][r] + bv;
            if (res)  v += res[(size_t)row * ldres + col];
            if (relu) v = fmaxf(v, 0.f);
            if (Cf) Cf[(size_t)row * ldcf + col] = v;
            if (Cb) Cb[(size_t)row * ldcb + col] = (bf16_t)v;
        }
    }
}

// ---------------------------------------------------------------------------
// Lw[n,g,m] = log(max(relu(pos_emb[n,m,:]·pos_w[g,:] + pos_b[g]), 1e-6))
// layout: Lw[(n*GRP+g)*MPAD + m]
// ---------------------------------------------------------------------------
__global__ __launch_bounds__(256)
void pos_logweight(const float* __restrict__ pos_emb,
                   const float* __restrict__ pw, const float* __restrict__ pb,
                   float* __restrict__ Lw)
{
    __shared__ float sw[GRP * EMBD];
    __shared__ float sb[GRP];
    for (int i = threadIdx.x; i < GRP * EMBD; i += 256) sw[i] = pw[i];
    if (threadIdx.x < GRP) sb[threadIdx.x] = pb[threadIdx.x];
    __syncthreads();

    int idx = blockIdx.x * 256 + threadIdx.x;     // (n, m)
    if (idx >= NTOT * NGT) return;
    int n = idx / NGT, m = idx - n * NGT;

    const float4* pe = (const float4*)(pos_emb + (size_t)idx * EMBD);
    float acc[GRP];
    #pragma unroll
    for (int g = 0; g < GRP; ++g) acc[g] = sb[g];
    for (int i = 0; i < EMBD / 4; ++i) {
        float4 e = pe[i];
        #pragma unroll
        for (int g = 0; g < GRP; ++g) {
            const float* w = &sw[g * EMBD + i * 4];
            acc[g] += e.x * w[0] + e.y * w[1] + e.z * w[2] + e.w * w[3];
        }
    }
    #pragma unroll
    for (int g = 0; g < GRP; ++g) {
        float a = fmaxf(acc[g], 0.f);
        Lw[((size_t)n * GRP + g) * MPAD + m] = logf(fmaxf(a, 1e-6f));
    }
}

// ---------------------------------------------------------------------------
// Row softmax over m<300 of (Lw + aff); writes bf16 s with zero padding to 320.
// One wave32 per (n,g) row.
// ---------------------------------------------------------------------------
__global__ __launch_bounds__(256)
void softmax_rows(const float* __restrict__ Lw, const float* __restrict__ aff,
                  bf16_t* __restrict__ s)
{
    int wid = threadIdx.x >> 5, lane = threadIdx.x & 31;
    int row = blockIdx.x * 8 + wid;               // [0, NTOT*GRP)
    size_t base = (size_t)row * MPAD;

    float v[10];
    float mx = -1e30f;
    #pragma unroll
    for (int i = 0; i < 10; ++i) {
        int m = lane + i * 32;
        float x = (m < NGT) ? (Lw[base + m] + aff[base + m]) : -1e30f;
        v[i] = x;
        mx = fmaxf(mx, x);
    }
    #pragma unroll
    for (int o = 16; o > 0; o >>= 1) mx = fmaxf(mx, __shfl_xor(mx, o, 32));
    float sum = 0.f;
    #pragma unroll
    for (int i = 0; i < 10; ++i) {
        int m = lane + i * 32;
        float e = (m < NGT) ? __expf(v[i] - mx) : 0.f;
        v[i] = e;
        sum += e;
    }
    #pragma unroll
    for (int o = 16; o > 0; o >>= 1) sum += __shfl_xor(sum, o, 32);
    float inv = 1.f / sum;
    #pragma unroll
    for (int i = 0; i < 10; ++i) {
        int m = lane + i * 32;
        s[base + m] = (bf16_t)(v[i] * inv);       // pad entries write exact 0
    }
}

__global__ void cvt_f32_bf16(const float* __restrict__ src, bf16_t* __restrict__ dst,
                             size_t n)
{
    size_t i = (size_t)blockIdx.x * 256 + threadIdx.x;
    if (i < n) dst[i] = (bf16_t)src[i];
}

// copy first 300 rows of h_bf16 (2048x1024) into 320x1024 buffer, zero pad rows
__global__ void make_nongt_pad(const bf16_t* __restrict__ hsrc, bf16_t* __restrict__ dst)
{
    int i = blockIdx.x * 256 + threadIdx.x;
    if (i >= MPAD * FEAT) return;
    int row = i >> 10;
    dst[i] = (row < NGT) ? hsrc[i] : (bf16_t)0.f;
}

// ---------------------------------------------------------------------------
static void rungemm(hipStream_t st,
                    const bf16_t* A, long batA, int lda,
                    const bf16_t* Bt, long batB, int ldb,
                    const float* bias, long batBias,
                    const float* res, long batRes, int ldres,
                    float* Cf, long batCf, int ldcf,
                    bf16_t* Cb, long batCb, int ldcb,
                    int M, int N, int K, int batch, float alpha, int relu)
{
    if (N % 128 == 0) {
        dim3 g((unsigned)(N / 128), (unsigned)(M / 64), (unsigned)batch), b(128);
        gemm_bf16_wmma<2><<<g, b, 0, st>>>(A, batA, lda, Bt, batB, ldb, bias, batBias,
                                           res, batRes, ldres, Cf, batCf, ldcf,
                                           Cb, batCb, ldcb, K, alpha, relu);
    } else {
        dim3 g((unsigned)(N / 64), (unsigned)(M / 64), (unsigned)batch), b(128);
        gemm_bf16_wmma<1><<<g, b, 0, st>>>(A, batA, lda, Bt, batB, ldb, bias, batBias,
                                           res, batRes, ldres, Cf, batCf, ldcf,
                                           Cb, batCb, ldcb, K, alpha, relu);
    }
}

extern "C" void kernel_launch(void* const* d_in, const int* in_sizes, int n_in,
                              void* d_out, int out_size, void* d_ws, size_t ws_size,
                              hipStream_t stream)
{
    (void)in_sizes; (void)n_in; (void)out_size; (void)ws_size;

    const float* x       = (const float*)d_in[0];
    const float* pos_emb = (const float*)d_in[1];
    const float* fc6_w   = (const float*)d_in[2];
    const float* fc6_b   = (const float*)d_in[3];
    const float* fc7_w   = (const float*)d_in[4];
    const float* fc7_b   = (const float*)d_in[5];
    // per attention stage: [pos_w,pos_b, q_w,q_b, k_w,k_b, out_w,out_b]
    const float* aw[2][4];
    const float* ab[2][4];
    for (int i = 0; i < 2; ++i) {
        int base = 6 + i * 8;
        for (int j = 0; j < 4; ++j) {
            aw[i][j] = (const float*)d_in[base + 2 * j];
            ab[i][j] = (const float*)d_in[base + 2 * j + 1];
        }
    }
    float* out = (float*)d_out;
    char*  ws  = (char*)d_ws;

    size_t off = 0;
    auto alloc = [&](size_t bytes) {
        size_t r = off;
        off += (bytes + 255) & ~(size_t)255;
        return r;
    };

    const size_t o_xb  = alloc((size_t)NTOT * INCH * 2);    // x bf16 (aliased by aff later)
    const size_t o_w6  = alloc((size_t)FEAT * INCH * 2);
    const size_t o_w7  = alloc((size_t)FEAT * FEAT * 2);
    size_t o_qw[2], o_kw[2], o_ow[2];
    for (int i = 0; i < 2; ++i) {
        o_qw[i] = alloc((size_t)FEAT * FEAT * 2);
        o_kw[i] = alloc((size_t)FEAT * FEAT * 2);
        o_ow[i] = alloc((size_t)FEAT * FEAT * 2);
    }
    const size_t o_h    = alloc((size_t)NTOT * FEAT * 4);   // fc6 out (aliased by fc7 out)
    const size_t o_hb   = alloc((size_t)NTOT * FEAT * 2);
    const size_t o_h1r  = alloc((size_t)NTOT * FEAT * 4);   // relu(h+att1)
    const size_t o_h1rb = alloc((size_t)NTOT * FEAT * 2);
    const size_t o_ng   = alloc((size_t)MPAD * FEAT * 2);   // nongt padded, bf16
    const size_t o_q    = alloc((size_t)NTOT * FEAT * 2);   // q bf16
    const size_t o_k    = alloc((size_t)MPAD * FEAT * 2);   // k bf16 (pad rows junk, unused)
    const size_t o_Lw   = alloc((size_t)NTOT * GRP * MPAD * 4);
    const size_t o_s    = alloc((size_t)NTOT * GRP * MPAD * 2);
    const size_t o_Pt   = alloc((size_t)GRP * DGRP * MPAD * 2);
    const size_t o_aff  = o_xb;   // alias: x_bf16 dead after fc6, aff (41.9MB) < xb (51.4MB)
    const size_t o_h2   = o_h;    // alias: h dead after attout1
    const size_t o_h2b  = o_hb;   // alias

    bf16_t* xb   = (bf16_t*)(ws + o_xb);
    bf16_t* w6b  = (bf16_t*)(ws + o_w6);
    bf16_t* w7b  = (bf16_t*)(ws + o_w7);
    bf16_t* qwb[2], *kwb[2], *owb[2];
    for (int i = 0; i < 2; ++i) {
        qwb[i] = (bf16_t*)(ws + o_qw[i]);
        kwb[i] = (bf16_t*)(ws + o_kw[i]);
        owb[i] = (bf16_t*)(ws + o_ow[i]);
    }
    float*  h    = (float*) (ws + o_h);
    bf16_t* hb   = (bf16_t*)(ws + o_hb);
    float*  h1r  = (float*) (ws + o_h1r);
    bf16_t* h1rb = (bf16_t*)(ws + o_h1rb);
    float*  h2   = (float*) (ws + o_h2);
    bf16_t* h2b  = (bf16_t*)(ws + o_h2b);
    bf16_t* ng   = (bf16_t*)(ws + o_ng);
    bf16_t* qbuf = (bf16_t*)(ws + o_q);
    bf16_t* kbuf = (bf16_t*)(ws + o_k);
    float*  Lw   = (float*) (ws + o_Lw);
    float*  aff  = (float*) (ws + o_aff);
    bf16_t* sbuf = (bf16_t*)(ws + o_s);
    bf16_t* Pt   = (bf16_t*)(ws + o_Pt);

    auto cvt = [&](const float* s_, bf16_t* d_, size_t n) {
        cvt_f32_bf16<<<dim3((unsigned)((n + 255) / 256)), dim3(256), 0, stream>>>(s_, d_, n);
    };

    // --- stage 0: convert inputs/weights to bf16 ---
    cvt(x, xb, (size_t)NTOT * INCH);
    cvt(fc6_w, w6b, (size_t)FEAT * INCH);
    cvt(fc7_w, w7b, (size_t)FEAT * FEAT);
    for (int i = 0; i < 2; ++i) {
        cvt(aw[i][1], qwb[i], (size_t)FEAT * FEAT);
        cvt(aw[i][2], kwb[i], (size_t)FEAT * FEAT);
        cvt(aw[i][3], owb[i], (size_t)FEAT * FEAT);
    }

    // --- fc6: h = x @ fc6_w^T + b ---
    rungemm(stream, xb, 0, INCH, w6b, 0, INCH, fc6_b, 0, nullptr, 0, 0,
            h, 0, FEAT, hb, 0, FEAT, NTOT, FEAT, INCH, 1, 1.f, 0);

    // --- relation attention (shared for both stages) ---
    auto attention = [&](const float* hf, const bf16_t* hbf, int i,
                         float* outF, bf16_t* outB) {
        // nongt = h[:300], zero-padded to 320 rows
        make_nongt_pad<<<dim3((MPAD * FEAT + 255) / 256), 256, 0, stream>>>(hbf, ng);
        // k = nongt @ k_w^T + k_b  (pad rows contaminated by bias; never read)
        rungemm(stream, ng, 0, FEAT, kwb[i], 0, FEAT, ab[i][2], 0, nullptr, 0, 0,
                nullptr, 0, 0, kbuf, 0, FEAT, MPAD, FEAT, FEAT, 1, 1.f, 0);
        // q = h @ q_w^T + q_b
        rungemm(stream, hbf, 0, FEAT, qwb[i], 0, FEAT, ab[i][1], 0, nullptr, 0, 0,
                nullptr, 0, 0, qbuf, 0, FEAT, NTOT, FEAT, FEAT, 1, 1.f, 0);
        // Lw[n,g,m] = log(max(relu(pos_emb·pos_w^T + b), 1e-6))
        pos_logweight<<<dim3((NTOT * NGT + 255) / 256), 256, 0, stream>>>(
            pos_emb, aw[i][0], ab[i][0], Lw);
        // aff[n,g,m] = (q_g · k_g) / 8 : batched over g (K=64)
        rungemm(stream, qbuf, DGRP, FEAT, kbuf, DGRP, FEAT, nullptr, 0, nullptr, 0, 0,
                aff, MPAD, GRP * MPAD, nullptr, 0, 0,
                NTOT, MPAD, DGRP, GRP, 0.125f, 0);
        // s = softmax(Lw + aff) over m<300, bf16, zero pad
        softmax_rows<<<dim3(NTOT * GRP / 8), 256, 0, stream>>>(Lw, aff, sbuf);
        // Pt[g,o,m] = out_w[g*64+o,:] · nongt[m,:]  (factored out_t einsum)
        rungemm(stream, owb[i], (long)DGRP * FEAT, FEAT, ng, 0, FEAT,
                nullptr, 0, nullptr, 0, 0, nullptr, 0, 0,
                Pt, (long)DGRP * MPAD, MPAD, DGRP, MPAD, FEAT, GRP, 1.f, 0);
        // out[:,g*64+o] = relu(h + s_g @ Pt_g^T + out_b)   (fused epilogue)
        rungemm(stream, sbuf, MPAD, GRP * MPAD, Pt, (long)DGRP * MPAD, MPAD,
                ab[i][3], DGRP, hf, DGRP, FEAT,
                outF, DGRP, FEAT, outB, DGRP, FEAT,
                NTOT, DGRP, MPAD, GRP, 1.f, 1);
    };

    // --- attention 1, fused h1 = relu(h + att1) ---
    attention(h, hb, 0, h1r, h1rb);

    // --- fc7: h2 = h1 @ fc7_w^T + b ---
    rungemm(stream, h1rb, 0, FEAT, w7b, 0, FEAT, fc7_b, 0, nullptr, 0, 0,
            h2, 0, FEAT, h2b, 0, FEAT, NTOT, FEAT, FEAT, 1, 1.f, 0);

    // --- attention 2, fused out = relu(h2 + att2) written straight to d_out ---
    attention(h2, h2b, 1, out, nullptr);
}
